// MultiHeadSelfAttention_53300544144106
// MI455X (gfx1250) — matrix operand order repaired and checked by reading, not compiled, not run on hardware
//
#include <hip/hip_runtime.h>

// Multi-head self-attention for gfx1250 (MI455X).
// bf16 WMMA (f32 accumulate) + flash softmax + async LDS staging of the
// tiles that are shared by all 8 waves of a workgroup (weights, K, V).
// B=4, S=2048, D=1024, H=16, d=64.

#define D_EMB 1024
#define NHEAD 16
#define HDIM  64
#define BATCH 4
#define SEQ   2048
#define NTOK  (BATCH * SEQ)

typedef __attribute__((ext_vector_type(16))) __bf16 v16bf;
typedef __attribute__((ext_vector_type(8)))  __bf16 v8bf;
typedef __attribute__((ext_vector_type(8)))  float  v8f;
typedef int v4i_vs __attribute__((vector_size(16)));  // matches builtin param

// ---------------------------------------------------------------------------
// CDNA5 async global->LDS copy (ASYNCcnt path), builtin if present, else asm.
// Probe result: builtin exists and takes (int4 AS1*, int4 AS3*, Ii, Ii).
// ---------------------------------------------------------------------------

static __device__ __forceinline__ void async_copy_b128(const __bf16* g,
                                                       __bf16* l) {
#if __has_builtin(__builtin_amdgcn_global_load_async_to_lds_b128)
  __builtin_amdgcn_global_load_async_to_lds_b128(
      (__attribute__((address_space(1))) v4i_vs*)(v4i_vs*)(void*)(__bf16*)g,
      (__attribute__((address_space(3))) v4i_vs*)(v4i_vs*)(void*)l,
      /*offset=*/0, /*cpol=*/0);
#else
  const unsigned loff =
      (unsigned)(__UINTPTR_TYPE__)(__attribute__((address_space(3))) char*)(void*)l;
  asm volatile("global_load_async_to_lds_b128 %0, %1, off"
               :: "v"(loff), "v"(g)
               : "memory");
#endif
}

static __device__ __forceinline__ void wait_async0() {
#if __has_builtin(__builtin_amdgcn_s_wait_asynccnt)
  __builtin_amdgcn_s_wait_asynccnt(0);
#else
  asm volatile("s_wait_asynccnt 0" ::: "memory");
#endif
}

// ---------------------------------------------------------------------------
// WMMA helpers (CDNA5 wave32 layouts, ISA 7.12.2)
// ---------------------------------------------------------------------------

static __device__ __forceinline__ v8f wmma_bf16(v16bf a, v16bf b, v8f c) {
  // D = A(16x32 bf16) * B(32x16 bf16) + C(16x16 f32)
  return __builtin_amdgcn_wmma_f32_16x16x32_bf16(
      /*neg_a=*/false, a, /*neg_b=*/false, b,
      /*c_mod=*/(short)0, c, /*reuse_a=*/false, /*reuse_b=*/false);
}

// A-matrix fragment: element (m,k) at base[(m0+m)*ld + k], row-major.
// Lane L holds row M = L%16.  lanes 0-15: K 0..15; lanes 16-31:
// K 16..31  -> two contiguous 16-byte halves per lane.
template <typename PTR>
static __device__ __forceinline__ v16bf load_a_frag(PTR base, int ld, int m0,
                                                    int kbase, int lane) {
  const int m   = m0 + (lane & 15);
  const int klo = (lane < 16) ? 0 : 16;
  auto p = base + (size_t)m * ld + kbase + klo;
  v8bf lo = *(const v8bf*)(p);
  v8bf hi = *(const v8bf*)(p + 8);
  v16bf r;
#pragma unroll
  for (int i = 0; i < 8; ++i) { r[i] = lo[i]; r[i + 8] = hi[i]; }
  return r;
}

// B-matrix fragment: element (k,n) at base[(n0+n)*ld + k]  (N-major source so
// the 16 consecutive K values per lane are one contiguous 32-byte load).
// Lane L holds column N = L%16; lanes 0-15: K 0..15, lanes 16-31: K 16..31.
template <typename PTR>
static __device__ __forceinline__ v16bf load_b_frag(PTR base, int ld, int n0,
                                                    int kbase, int lane) {
  const int n  = n0 + (lane & 15);
  const int kb = (lane < 16) ? 0 : 16;
  return *(const v16bf*)(base + (size_t)n * ld + kbase + kb);
}

// ---------------------------------------------------------------------------
// Precision / layout conversion kernels
// ---------------------------------------------------------------------------

__global__ void cvt_f32_bf16(const float* __restrict__ in,
                             __bf16* __restrict__ out, int n) {
  int i = blockIdx.x * blockDim.x + threadIdx.x;
  if (i < n) out[i] = (__bf16)in[i];
}

// W[k][n] (fp32, DxD) -> Wt[n][k] (bf16), LDS-tiled transpose for coalescing.
__global__ void transpose_w_bf16(const float* __restrict__ W,
                                 __bf16* __restrict__ Wt) {
  __shared__ float tile[16][17];
  const int n = blockIdx.x * 16 + threadIdx.x;
  const int k = blockIdx.y * 16 + threadIdx.y;
  tile[threadIdx.y][threadIdx.x] = W[(size_t)k * D_EMB + n];
  __syncthreads();
  const int nt = blockIdx.x * 16 + threadIdx.y;
  const int kt = blockIdx.y * 16 + threadIdx.x;
  Wt[(size_t)nt * D_EMB + kt] = (__bf16)tile[threadIdx.x][threadIdx.y];
}

// ---------------------------------------------------------------------------
// GEMM:  C(8192 x 1024) = A(8192 x 1024, bf16 row-major) * Bt^T + bias
// Bt is N-major (Bt[n][k]).  8 waves per block; all waves share the same
// 64-wide N strip, so the Bt tile (64x32 = 4KB) is staged once per K-step
// into LDS with async copies and double buffering.  Epilogue by MODE:
//   0: bf16 -> (B,H,S,d)                  (K)
//   3: bf16 -> (B,H,S,d), scaled 1/sqrt(d) (Q, folds softmax scale)
//   1: bf16 -> (B,H,d,S)                  (V transposed for PV B-frags)
//   2: fp32 row-major -> d_out            (output projection)
// ---------------------------------------------------------------------------

template <int MODE>
__global__ __launch_bounds__(256) void gemm_kernel(
    const __bf16* __restrict__ A, const __bf16* __restrict__ Bt,
    const float* __restrict__ bias, __bf16* __restrict__ out_bf,
    float* __restrict__ out_f32) {
  __shared__ __bf16 bsm[2][64 * 32];  // [n][k] N-major tile, double buffered

  const int tid  = threadIdx.x;
  const int lane = tid & 31;
  const int wave = tid >> 5;
  const int m0 = (blockIdx.x * 8 + wave) * 16;
  const int n0 = blockIdx.y * 64;

  // cooperative async stage of Bt[n0..n0+64) x [kb..kb+32): 16B per thread
  const int sn = tid >> 2, sseg = (tid & 3) * 8;
  const __bf16* bsrc = Bt + (size_t)(n0 + sn) * D_EMB + sseg;

  async_copy_b128(bsrc + 0, &bsm[0][sn * 32 + sseg]);  // preload tile 0

  v8f acc[4] = {};

  for (int it = 0; it < D_EMB / 32; ++it) {
    const int kb = it * 32;
    const int buf = it & 1;
    wait_async0();
    __syncthreads();
    if (kb + 32 < D_EMB)
      async_copy_b128(bsrc + kb + 32, &bsm[buf ^ 1][sn * 32 + sseg]);

    __builtin_prefetch(A + (size_t)(m0 + (lane & 15)) * D_EMB + kb + 64, 0, 0);
    const v16bf a = load_a_frag(A, D_EMB, m0, kb, lane);
#pragma unroll
    for (int t = 0; t < 4; ++t) {
      const v16bf b = load_b_frag(&bsm[buf][0], 32, t * 16, 0, lane);
      acc[t] = wmma_bf16(a, b, acc[t]);
    }
  }

  const int half8 = (lane >> 4) << 3;  // C layout: row = m0 + r + 8*(lane>=16)
#pragma unroll
  for (int t = 0; t < 4; ++t) {
    const int col = n0 + t * 16 + (lane & 15);
    const float bv = bias[col];
#pragma unroll
    for (int r = 0; r < 8; ++r) {
      const int tok = m0 + r + half8;
      float val = acc[t][r] + bv;
      if (MODE == 3) val *= 0.125f;  // 1/sqrt(HDIM), folded into Q
      if (MODE == 0 || MODE == 3) {
        const int b = tok >> 11, s = tok & (SEQ - 1);
        const int h = col >> 6, dd = col & 63;
        out_bf[(((size_t)(b * NHEAD + h) * SEQ + s) << 6) + dd] = (__bf16)val;
      } else if (MODE == 1) {
        const int b = tok >> 11, s = tok & (SEQ - 1);
        const int h = col >> 6, dd = col & 63;
        out_bf[((size_t)(b * NHEAD + h) * HDIM + dd) * SEQ + s] = (__bf16)val;
      } else {
        out_f32[(size_t)tok * D_EMB + col] = val;
      }
    }
  }
}

// ---------------------------------------------------------------------------
// Flash attention: 8 waves per block, all on the same (b,h); each wave owns a
// 16-row Q tile.  K/V tiles (32 keys) are shared by all waves -> staged into
// LDS with async copies, double buffered.  Scores never touch HBM.
// ---------------------------------------------------------------------------

__global__ __launch_bounds__(256) void attn_kernel(
    const __bf16* __restrict__ q,   // (B,H,S,d), pre-scaled by 1/sqrt(d)
    const __bf16* __restrict__ k,   // (B,H,S,d)
    const __bf16* __restrict__ v,   // (B,H,d,S)
    __bf16* __restrict__ attn) {    // (B,S,D) concat layout
  __shared__ __bf16 ksm[2][32 * 64];  // [key][dim], 4KB per buffer
  __shared__ __bf16 vsm[2][64 * 32];  // [dim][key], 4KB per buffer
  __shared__ __bf16 psm[8 * 16 * 32]; // per-wave 16x32 P tile

  const int tid  = threadIdx.x;
  const int lane = tid & 31;
  const int wave = tid >> 5;
  __bf16* pt = psm + wave * (16 * 32);

  const int bh = blockIdx.y;                     // b*H + h
  const int s0 = (blockIdx.x * 8 + wave) * 16;   // query rows [s0, s0+16)

  const __bf16* qb = q + (size_t)bh * SEQ * HDIM;
  const __bf16* kbp = k + (size_t)bh * SEQ * HDIM;
  const __bf16* vbp = v + (size_t)bh * HDIM * SEQ;

  // cooperative stage addressing: 16B per thread per tile
  const int krow = tid >> 3, kseg = (tid & 7) * 8;   // K: 32 rows x 64 dims
  const int vrow = tid >> 2, vseg = (tid & 3) * 8;   // V: 64 dims x 32 keys

  // preload key-tile 0
  async_copy_b128(kbp + (size_t)krow * HDIM + kseg, &ksm[0][krow * 64 + kseg]);
  async_copy_b128(vbp + (size_t)vrow * SEQ + vseg,  &vsm[0][vrow * 32 + vseg]);

  // Q fragments for the whole head dim (K = 0..63 in two 32-chunks)
  const v16bf aq0 = load_a_frag(qb, HDIM, s0, 0, lane);
  const v16bf aq1 = load_a_frag(qb, HDIM, s0, 32, lane);

  float mrow[8], lrow[8];
  v8f acc[4] = {};
#pragma unroll
  for (int r = 0; r < 8; ++r) { mrow[r] = -3.0e38f; lrow[r] = 0.0f; }

  const int half8 = (lane >> 4) << 3;
  const int cn = lane & 15;

  for (int sk = 0; sk < SEQ; sk += 32) {
    const int buf = (sk >> 5) & 1;
    wait_async0();
    __syncthreads();
    if (sk + 32 < SEQ) {
      async_copy_b128(kbp + (size_t)(sk + 32 + krow) * HDIM + kseg,
                      &ksm[buf ^ 1][krow * 64 + kseg]);
      async_copy_b128(vbp + (size_t)vrow * SEQ + sk + 32 + vseg,
                      &vsm[buf ^ 1][vrow * 32 + vseg]);
    }

    // ---- scores: S = Qs * K^T (Q pre-scaled), two 16x16 fragments ----
    v8f sc[2] = {};
#pragma unroll
    for (int nt = 0; nt < 2; ++nt) {
      const v16bf b0 = load_b_frag(&ksm[buf][0], 64, nt * 16, 0, lane);
      const v16bf b1 = load_b_frag(&ksm[buf][0], 64, nt * 16, 32, lane);
      sc[nt] = wmma_bf16(aq0, b0, sc[nt]);
      sc[nt] = wmma_bf16(aq1, b1, sc[nt]);
    }

    // ---- online softmax (row stats live in the 16-lane half owning M) ----
#pragma unroll
    for (int r = 0; r < 8; ++r) {
      float mt = fmaxf(sc[0][r], sc[1][r]);
#pragma unroll
      for (int off = 1; off < 16; off <<= 1)
        mt = fmaxf(mt, __shfl_xor(mt, off, 32));
      const float mnew = fmaxf(mrow[r], mt);
      const float fac = __expf(mrow[r] - mnew);
      const float p0 = __expf(sc[0][r] - mnew);
      const float p1 = __expf(sc[1][r] - mnew);
      sc[0][r] = p0;
      sc[1][r] = p1;
      float rs = p0 + p1;
#pragma unroll
      for (int off = 1; off < 16; off <<= 1) rs += __shfl_xor(rs, off, 32);
      lrow[r] = lrow[r] * fac + rs;
      mrow[r] = mnew;
#pragma unroll
      for (int t = 0; t < 4; ++t) acc[t][r] *= fac;
    }

    // ---- re-stage P: C layout -> LDS (16x32 row-major) -> A layout ----
#pragma unroll
    for (int f = 0; f < 2; ++f)
#pragma unroll
      for (int r = 0; r < 8; ++r)
        pt[(r + half8) * 32 + f * 16 + cn] = (__bf16)sc[f][r];
    __builtin_amdgcn_wave_barrier();
    const v16bf pA = load_a_frag((const __bf16*)pt, 32, 0, 0, lane);
    __builtin_amdgcn_wave_barrier();

    // ---- accumulate P(16x32) * V(32x64) across 4 head-dim tiles ----
#pragma unroll
    for (int t = 0; t < 4; ++t) {
      const v16bf bvf = load_b_frag(&vsm[buf][0], 32, t * 16, 0, lane);
      acc[t] = wmma_bf16(pA, bvf, acc[t]);
    }
  }

  // ---- epilogue: normalize and scatter into (B,S,H,d) = (B,S,D) concat ----
  const int b = bh / NHEAD, h = bh % NHEAD;
#pragma unroll
  for (int r = 0; r < 8; ++r) {
    const float inv = 1.0f / lrow[r];
    const int srow = s0 + r + half8;
#pragma unroll
    for (int t = 0; t < 4; ++t) {
      const size_t idx =
          (size_t)(b * SEQ + srow) * D_EMB + h * HDIM + t * 16 + cn;
      attn[idx] = (__bf16)(acc[t][r] * inv);
    }
  }
}

// ---------------------------------------------------------------------------
// Host-side orchestration
// ---------------------------------------------------------------------------

extern "C" void kernel_launch(void* const* d_in, const int* in_sizes, int n_in,
                              void* d_out, int out_size, void* d_ws,
                              size_t ws_size, hipStream_t stream) {
  const float* x  = (const float*)d_in[0];
  const float* Wq = (const float*)d_in[1];
  const float* bq = (const float*)d_in[2];
  const float* Wk = (const float*)d_in[3];
  const float* bk = (const float*)d_in[4];
  const float* Wv = (const float*)d_in[5];
  const float* bv = (const float*)d_in[6];
  const float* Wo = (const float*)d_in[7];
  const float* bo = (const float*)d_in[8];
  float* out = (float*)d_out;

  // Workspace layout (bytes): ~88 MB total.
  char* ws = (char*)d_ws;
  const size_t xbytes = (size_t)NTOK * D_EMB * 2;           // 16 MB
  const size_t wbytes = (size_t)D_EMB * D_EMB * 2;          //  2 MB each
  __bf16* x_bf    = (__bf16*)(ws);
  __bf16* wqT     = (__bf16*)(ws + xbytes);
  __bf16* wkT     = (__bf16*)(ws + xbytes + wbytes);
  __bf16* wvT     = (__bf16*)(ws + xbytes + 2 * wbytes);
  __bf16* woT     = (__bf16*)(ws + xbytes + 3 * wbytes);
  __bf16* q_bf    = (__bf16*)(ws + xbytes + 4 * wbytes);
  __bf16* k_bf    = (__bf16*)(ws + 2 * xbytes + 4 * wbytes);
  __bf16* v_bf    = (__bf16*)(ws + 3 * xbytes + 4 * wbytes);
  __bf16* attn_bf = (__bf16*)(ws + 4 * xbytes + 4 * wbytes);

  // 1) input -> bf16
  const int nelem = NTOK * D_EMB;
  cvt_f32_bf16<<<(nelem + 255) / 256, 256, 0, stream>>>(x, x_bf, nelem);

  // 2) weights -> bf16, N-major (transposed)
  dim3 tb(16, 16), tg(D_EMB / 16, D_EMB / 16);
  transpose_w_bf16<<<tg, tb, 0, stream>>>(Wq, wqT);
  transpose_w_bf16<<<tg, tb, 0, stream>>>(Wk, wkT);
  transpose_w_bf16<<<tg, tb, 0, stream>>>(Wv, wvT);
  transpose_w_bf16<<<tg, tb, 0, stream>>>(Wo, woT);

  // 3) QKV projections (WMMA + async-staged weight tiles)
  dim3 gg(NTOK / 16 / 8, D_EMB / 64);
  gemm_kernel<3><<<gg, 256, 0, stream>>>(x_bf, wqT, bq, q_bf, nullptr);
  gemm_kernel<0><<<gg, 256, 0, stream>>>(x_bf, wkT, bk, k_bf, nullptr);
  gemm_kernel<1><<<gg, 256, 0, stream>>>(x_bf, wvT, bv, v_bf, nullptr);

  // 4) flash attention (WMMA scores + WMMA PV, async-staged K/V tiles)
  attn_kernel<<<dim3(SEQ / 16 / 8, BATCH * NHEAD), 256, 0, stream>>>(
      q_bf, k_bf, v_bf, attn_bf);

  // 5) output projection -> fp32 d_out
  gemm_kernel<2><<<gg, 256, 0, stream>>>(attn_bf, woT, bo, nullptr, out);
}
